// JRegularizedCrossEntropyLoss_4733053960214
// MI455X (gfx1250) — matrix operand order repaired and checked by reading, not compiled, not run on hardware
//
#include <hip/hip_runtime.h>
#include <hip/hip_bf16.h>

// MI455X (gfx1250) single-pass implementation.
//
// Pass 1 (main_kernel): stream pred once through LDS. From the LDS staging:
//   - waves run v_wmma_f32_16x16x32_f16 with A = [pred rows 0..7 ; ones rows 8..15]
//     and B = onehot(target==col)  => D rows 0..7 = S[b,ci,ck], D rows 8..15 = n[b,ck]
//   - threads compute per-pixel 8-way logsumexp NLL for cross-entropy
//   Partials pushed with f32 global atomics into workspace.
// Pass 2 (finalize_kernel): 1 block computes M=S/n, the J log terms, and jl+ce.
//
// Memory-bound: ~150MB read once => ~6.5us floor at 23.3 TB/s.
// Staging uses GLOBAL_LOAD_ASYNC_TO_LDS_B128 (ASYNCcnt-tracked direct-to-LDS DMA)
// when the toolchain exposes the builtin; falls back to load+ds_store otherwise.

typedef __attribute__((ext_vector_type(16))) _Float16 v16h;
typedef __attribute__((ext_vector_type(8)))  float    v8f;
typedef __attribute__((ext_vector_type(4)))  int      i4v;
typedef __attribute__((address_space(1))) i4v *gi4p;   // global int4*
typedef __attribute__((address_space(3))) i4v *li4p;   // LDS int4*

#define BQ   16
#define CQ   8
#define HWQ  262144          // 512*512
#define TQ   1024            // pixels per block chunk
#define PSTR 1028            // LDS row stride in floats (1028 % 64 == 4 -> bank spread)
#define NTHR 256

#if __has_builtin(__builtin_amdgcn_global_load_async_to_lds_b128) && \
    __has_builtin(__builtin_amdgcn_s_wait_asynccnt)
#define USE_ASYNC_LDS 1
#else
#define USE_ASYNC_LDS 0
#endif

__global__ __launch_bounds__(NTHR)
void jce_main_kernel(const float* __restrict__ pred,
                     const int*   __restrict__ tgt,   // int32 per bridge doc
                     float* __restrict__ gS,          // [B][8][8]
                     float* __restrict__ gC,          // [B][8]
                     float* __restrict__ gCE)         // [1]
{
    __shared__ float sP[CQ * PSTR];   // 8 x 1028 floats, padded rows
    __shared__ int   sT[TQ];
    __shared__ float sS[64];
    __shared__ float sCnt[8];
    __shared__ float sRed[NTHR];

    const int t    = threadIdx.x;
    const int b    = blockIdx.y;
    const int chnk = blockIdx.x;             // 0..255
    const size_t pbase = (size_t)b * CQ * HWQ + (size_t)chnk * TQ;
    const size_t tbase = (size_t)b * HWQ + (size_t)chnk * TQ;

    if (t < 64) sS[t]   = 0.0f;
    if (t < 8)  sCnt[t] = 0.0f;

    // ---- stage chunk into LDS (16B per lane per transfer, fully coalesced) ----
#if USE_ASYNC_LDS
    #pragma unroll
    for (int c = 0; c < CQ; ++c) {
        __builtin_amdgcn_global_load_async_to_lds_b128(
            (gi4p)(pred + pbase + (size_t)c * HWQ + t * 4),
            (li4p)&sP[c * PSTR + t * 4],
            0, 0);
    }
    __builtin_amdgcn_global_load_async_to_lds_b128(
        (gi4p)(tgt + tbase + t * 4),
        (li4p)&sT[t * 4],
        0, 0);
    __builtin_amdgcn_s_wait_asynccnt(0);   // this wave's transfers done; barrier below
#else
    #pragma unroll
    for (int c = 0; c < CQ; ++c) {
        float4 v = *(const float4*)(pred + pbase + (size_t)c * HWQ + t * 4);
        *(float4*)&sP[c * PSTR + t * 4] = v;
    }
    {
        int4 tv = *(const int4*)(tgt + tbase + t * 4);
        *(int4*)&sT[t * 4] = tv;
    }
#endif
    __syncthreads();

    // ---- WMMA phase: S[b] (8x8) + counts via ones-rows, 32 pixels per wmma ----
    const int lane = t & 31;
    const int wv   = t >> 5;          // 8 waves, 128 pixels each
    const int m    = lane & 15;       // A row / B column for this lane
    const int h    = lane >> 4;       // half selector for K mapping

    v8f acc = {0.f, 0.f, 0.f, 0.f, 0.f, 0.f, 0.f, 0.f};

    #pragma unroll
    for (int k = 0; k < 4; ++k) {
        const int p0 = wv * 128 + k * 32;

        // A: lanes h=0 hold K=0-7 & 16-23, h=1 hold K=8-15 & 24-31  (pixel = p0+K)
        v16h A;
        if (m < CQ) {
            const float* row = &sP[m * PSTR + p0 + h * 8];
            float4 x0 = *(const float4*)(row);
            float4 x1 = *(const float4*)(row + 4);
            float4 x2 = *(const float4*)(row + 16);
            float4 x3 = *(const float4*)(row + 20);
            A[0]=(_Float16)x0.x; A[1]=(_Float16)x0.y; A[2]=(_Float16)x0.z; A[3]=(_Float16)x0.w;
            A[4]=(_Float16)x1.x; A[5]=(_Float16)x1.y; A[6]=(_Float16)x1.z; A[7]=(_Float16)x1.w;
            A[8]=(_Float16)x2.x; A[9]=(_Float16)x2.y; A[10]=(_Float16)x2.z; A[11]=(_Float16)x2.w;
            A[12]=(_Float16)x3.x; A[13]=(_Float16)x3.y; A[14]=(_Float16)x3.z; A[15]=(_Float16)x3.w;
        } else {
            // ones rows 8..15 -> D rows 8..15 accumulate per-class pixel counts
            #pragma unroll
            for (int j = 0; j < 16; ++j) A[j] = (_Float16)1.0f;
        }

        // B: lane's column n=m; h=0 holds K=0-7 & 16-23, h=1 holds K=8-15 & 24-31
        const int* tp = &sT[p0 + h * 8];
        int4 t0 = *(const int4*)(tp);
        int4 t1 = *(const int4*)(tp + 4);
        int4 t2 = *(const int4*)(tp + 16);
        int4 t3 = *(const int4*)(tp + 20);
        v16h Bv;
        Bv[0] =(_Float16)((t0.x==m)?1.0f:0.0f); Bv[1] =(_Float16)((t0.y==m)?1.0f:0.0f);
        Bv[2] =(_Float16)((t0.z==m)?1.0f:0.0f); Bv[3] =(_Float16)((t0.w==m)?1.0f:0.0f);
        Bv[4] =(_Float16)((t1.x==m)?1.0f:0.0f); Bv[5] =(_Float16)((t1.y==m)?1.0f:0.0f);
        Bv[6] =(_Float16)((t1.z==m)?1.0f:0.0f); Bv[7] =(_Float16)((t1.w==m)?1.0f:0.0f);
        Bv[8] =(_Float16)((t2.x==m)?1.0f:0.0f); Bv[9] =(_Float16)((t2.y==m)?1.0f:0.0f);
        Bv[10]=(_Float16)((t2.z==m)?1.0f:0.0f); Bv[11]=(_Float16)((t2.w==m)?1.0f:0.0f);
        Bv[12]=(_Float16)((t3.x==m)?1.0f:0.0f); Bv[13]=(_Float16)((t3.y==m)?1.0f:0.0f);
        Bv[14]=(_Float16)((t3.z==m)?1.0f:0.0f); Bv[15]=(_Float16)((t3.w==m)?1.0f:0.0f);

        // D = A*B + acc  (EXEC all-ones here; divergence above has re-converged)
        acc = __builtin_amdgcn_wmma_f32_16x16x32_f16(
                  false, A, false, Bv, (short)0, acc, false, false);
    }

    // extract: lanes 0..7 vgpr r -> S[r][lane]; lanes 16..23 vgpr 0 -> counts[lane-16]
    if (lane < 8) {
        #pragma unroll
        for (int r = 0; r < 8; ++r) atomicAdd(&sS[r * 8 + lane], acc[r]);
    } else if (lane >= 16 && lane < 24) {
        atomicAdd(&sCnt[lane - 16], acc[0]);
    }

    // ---- CE phase: per-pixel 8-way logsumexp NLL from the same LDS staging ----
    float ce = 0.0f;
    #pragma unroll
    for (int i = 0; i < TQ / NTHR; ++i) {
        const int p = t + i * NTHR;
        float x0 = sP[0*PSTR + p], x1 = sP[1*PSTR + p];
        float x2 = sP[2*PSTR + p], x3 = sP[3*PSTR + p];
        float x4 = sP[4*PSTR + p], x5 = sP[5*PSTR + p];
        float x6 = sP[6*PSTR + p], x7 = sP[7*PSTR + p];
        float mx = fmaxf(fmaxf(fmaxf(x0,x1),fmaxf(x2,x3)),
                         fmaxf(fmaxf(x4,x5),fmaxf(x6,x7)));
        float s = __expf(x0-mx) + __expf(x1-mx) + __expf(x2-mx) + __expf(x3-mx)
                + __expf(x4-mx) + __expf(x5-mx) + __expf(x6-mx) + __expf(x7-mx);
        float lse = mx + __logf(s);
        int   tv  = sT[p];
        float xt  = sP[tv * PSTR + p];
        ce += lse - xt;
    }
    sRed[t] = ce;
    __syncthreads();
    #pragma unroll
    for (int o = NTHR / 2; o > 0; o >>= 1) {
        if (t < o) sRed[t] += sRed[t + o];
        __syncthreads();
    }

    // ---- push block partials ----
    if (t == 0)  atomicAdd(gCE, sRed[0]);
    if (t < 64)  atomicAdd(&gS[b * 64 + t], sS[t]);
    if (t < 8)   atomicAdd(&gC[b * 8 + t], sCnt[t]);
}

__global__ __launch_bounds__(1024)
void jce_finalize_kernel(const float* __restrict__ gS,
                         const float* __restrict__ gC,
                         const float* __restrict__ gCE,
                         float* __restrict__ out)
{
    __shared__ float red[1024];
    const int t  = threadIdx.x;          // 16*8*8 == 1024 exactly
    const int b  = t >> 6;
    const int ci = (t >> 3) & 7;
    const int ck = t & 7;

    float Mik  = gS[b * 64 + ci * 8 + ck] / gC[b * 8 + ck];
    float diag = gS[b * 64 + ci * 8 + ci] / gC[b * 8 + ci];
    float term = (ci != ck) ? __logf(0.5f + 0.5f * (diag - Mik)) : 0.0f;

    red[t] = term;
    __syncthreads();
    #pragma unroll
    for (int o = 512; o > 0; o >>= 1) {
        if (t < o) red[t] += red[t + o];
        __syncthreads();
    }
    if (t == 0) {
        float jl = -red[0] / (float)BQ;
        float ce = gCE[0] / ((float)BQ * (float)HWQ);
        out[0] = jl + ce;
    }
}

__global__ void jce_zero_ws(float* ws, int n)
{
    int i = blockIdx.x * blockDim.x + threadIdx.x;
    if (i < n) ws[i] = 0.0f;
}

extern "C" void kernel_launch(void* const* d_in, const int* in_sizes, int n_in,
                              void* d_out, int out_size, void* d_ws, size_t ws_size,
                              hipStream_t stream) {
    (void)in_sizes; (void)n_in; (void)out_size; (void)ws_size;
    const float* pred = (const float*)d_in[0];
    const int*   tgt  = (const int*)d_in[1];     // target as int32 (bridge doc)
    float*       out  = (float*)d_out;

    float* ws  = (float*)d_ws;
    float* gS  = ws;               // 16*64 = 1024 floats
    float* gC  = ws + 1024;        // 16*8  = 128 floats
    float* gCE = ws + 1152;        // 1 float

    jce_zero_ws<<<5, 256, 0, stream>>>(ws, 1153);
    jce_main_kernel<<<dim3(HWQ / TQ, BQ), NTHR, 0, stream>>>(pred, tgt, gS, gC, gCE);
    jce_finalize_kernel<<<1, 1024, 0, stream>>>(gS, gC, gCE, out);
}
